// CausalSelfAttention_88149908783299
// MI455X (gfx1250) — matrix operand order repaired and checked by reading, not compiled, not run on hardware
//
#include <hip/hip_runtime.h>
#include <hip/hip_bf16.h>

// Causal self-attention, B=4 T=2048 C=1024 H=16 D=64, fp32 I/O.
// Strategy: convert to f16 once (everything is L2-resident on MI455X's 192MB L2),
// then run QKV GEMMs, flash attention, and output GEMM entirely on
// v_wmma_f32_16x16x32_f16 (wave32 WMMA, fp32 accumulate).
// GEMM wave tile = 32x64 (8 accumulators): 8 WMMAs per 12 b128 loads.
// Workspace requirement: 88 MB.

typedef __attribute__((ext_vector_type(16))) _Float16 v16h;
typedef __attribute__((ext_vector_type(8)))  _Float16 v8h;
typedef __attribute__((ext_vector_type(4)))  _Float16 v4h;
typedef __attribute__((ext_vector_type(8)))  float    v8f;

#define BATT 4
#define TSEQ 2048
#define CDIM 1024
#define NH   16
#define DH   64

__device__ __forceinline__ v8f wmma_f16(v16h a, v16h b, v8f c) {
  // (neg_a, A, neg_b, B, c_mod, C, reuse_a, reuse_b)
  return __builtin_amdgcn_wmma_f32_16x16x32_f16(false, a, false, b, (short)0, c,
                                                false, false);
}

// A-matrix 16x32 f16 fragment (MxK), row-major source with leading dim ld.
// Lane layout: row = lane%16; halves [0..7] = K at (hi*8)..(hi*8+7),
// halves [8..15] = K at (hi*8+16)..(hi*8+23), hi = lane/16.
__device__ __forceinline__ v16h load_a_frag(const _Float16* __restrict__ base,
                                            int ld) {
  int lane = threadIdx.x & 31;
  const _Float16* p = base + (size_t)(lane & 15) * ld + ((lane >> 4) << 3);
  v8h lo = *(const v8h*)p;
  v8h hi = *(const v8h*)(p + 16);
  return __builtin_shufflevector(lo, hi, 0, 1, 2, 3, 4, 5, 6, 7, 8, 9, 10, 11,
                                 12, 13, 14, 15);
}

// B-matrix 32x16 f16 fragment (KxN) in NT form: column n of B = row n of
// 'base' (row-major, leading dim ld). Lane layout: col = lane%16,
// halves [0..7] = K at (hi*8)..(hi*8+7), [8..15] = K at (hi*8+16)..(hi*8+23).
__device__ __forceinline__ v16h load_b_frag(const _Float16* __restrict__ base,
                                            int ld) {
  int lane = threadIdx.x & 31;
  const _Float16* p = base + (size_t)(lane & 15) * ld + ((lane >> 4) << 3);
  v8h lo = *(const v8h*)p;
  v8h hi = *(const v8h*)(p + 16);
  return __builtin_shufflevector(lo, hi, 0, 1, 2, 3, 4, 5, 6, 7, 8, 9, 10, 11,
                                 12, 13, 14, 15);
}

__device__ __forceinline__ float rmax16(float x) {
  x = fmaxf(x, __shfl_xor(x, 1));
  x = fmaxf(x, __shfl_xor(x, 2));
  x = fmaxf(x, __shfl_xor(x, 4));
  x = fmaxf(x, __shfl_xor(x, 8));
  return x;
}
__device__ __forceinline__ float rsum16(float x) {
  x += __shfl_xor(x, 1);
  x += __shfl_xor(x, 2);
  x += __shfl_xor(x, 4);
  x += __shfl_xor(x, 8);
  return x;
}

// ---------------- f32 -> f16 conversion (4 elems/thread) ----------------
__global__ __launch_bounds__(256) void cvt_f32_f16_kernel(
    const float* __restrict__ in, _Float16* __restrict__ out, int n) {
  int i = (blockIdx.x * blockDim.x + threadIdx.x) * 4;
  if (i < n) {
    float4 f = *(const float4*)(in + i);
    v4h h;
    h[0] = (_Float16)f.x;
    h[1] = (_Float16)f.y;
    h[2] = (_Float16)f.z;
    h[3] = (_Float16)f.w;
    *(v4h*)(out + i) = h;
  }
}

// --------------- shared GEMM core: 32x64 wave tile, K=1024 ---------------
// Computes acc[2][4] for out tile rows [row0,row0+32), cols [col0,col0+64).
__device__ __forceinline__ void gemm_core_32x64(const _Float16* __restrict__ X,
                                                const _Float16* __restrict__ W,
                                                int row0, int col0,
                                                v8f acc[2][4]) {
  const int K = CDIM;
  const _Float16* xa = X + (size_t)row0 * K;
  const _Float16* wb = W + (size_t)col0 * K;
  for (int kk = 0; kk < K; kk += 32) {
    __builtin_prefetch(xa + kk + 256, 0, 3);  // -> global_prefetch_b8
    __builtin_prefetch(wb + kk + 256, 0, 3);
    v16h a0 = load_a_frag(xa + kk, K);
    v16h a1 = load_a_frag(xa + (size_t)16 * K + kk, K);
    v16h b0 = load_b_frag(wb + kk, K);
    v16h b1 = load_b_frag(wb + (size_t)16 * K + kk, K);
    v16h b2 = load_b_frag(wb + (size_t)32 * K + kk, K);
    v16h b3 = load_b_frag(wb + (size_t)48 * K + kk, K);
    acc[0][0] = wmma_f16(a0, b0, acc[0][0]);
    acc[0][1] = wmma_f16(a0, b1, acc[0][1]);
    acc[0][2] = wmma_f16(a0, b2, acc[0][2]);
    acc[0][3] = wmma_f16(a0, b3, acc[0][3]);
    acc[1][0] = wmma_f16(a1, b0, acc[1][0]);
    acc[1][1] = wmma_f16(a1, b1, acc[1][1]);
    acc[1][2] = wmma_f16(a1, b2, acc[1][2]);
    acc[1][3] = wmma_f16(a1, b3, acc[1][3]);
  }
}

// ---------------- QKV projection GEMM: out = x @ W^T + b ----------------
// M=8192 (B*T), N=1024 (C), K=1024 (C). 8 waves/block, wave tile 32x64,
// block tile 128x128. MODE 0: out f16 [B,H,T,D]; MODE 1: out f16 [B,H,D,T].
template <int MODE>
__global__ __launch_bounds__(256) void proj_qkv_kernel(
    const _Float16* __restrict__ X, const _Float16* __restrict__ W,
    const float* __restrict__ bias, _Float16* __restrict__ out) {
  int wid = threadIdx.x >> 5, lane = threadIdx.x & 31;
  int row0 = blockIdx.x * 128 + (wid & 3) * 32;
  int col0 = blockIdx.y * 128 + (wid >> 2) * 64;
  v8f acc[2][4] = {};
  gemm_core_32x64(X, W, row0, col0, acc);
  int hi = lane >> 4;
#pragma unroll
  for (int mt = 0; mt < 2; ++mt) {
#pragma unroll
    for (int nt = 0; nt < 4; ++nt) {
      int n = col0 + nt * 16 + (lane & 15);
      float bn = bias[n];
      int h = n >> 6, d = n & 63;  // D = 64
#pragma unroll
      for (int v = 0; v < 8; ++v) {
        int m = row0 + mt * 16 + hi * 8 + v;
        int b = m >> 11, t = m & 2047;  // T = 2048
        float val = acc[mt][nt][v] + bn;
        if (MODE == 0)
          out[(((size_t)(b * NH + h) * TSEQ) + t) * DH + d] = (_Float16)val;
        else
          out[(((size_t)(b * NH + h) * DH) + d) * TSEQ + t] = (_Float16)val;
      }
    }
  }
}

// ------------- output projection GEMM: out = y @ Wp^T + bp (fp32) -------------
__global__ __launch_bounds__(256) void proj_out_kernel(
    const _Float16* __restrict__ X, const _Float16* __restrict__ W,
    const float* __restrict__ bias, float* __restrict__ out) {
  int wid = threadIdx.x >> 5, lane = threadIdx.x & 31;
  int row0 = blockIdx.x * 128 + (wid & 3) * 32;
  int col0 = blockIdx.y * 128 + (wid >> 2) * 64;
  v8f acc[2][4] = {};
  gemm_core_32x64(X, W, row0, col0, acc);
  int hi = lane >> 4;
#pragma unroll
  for (int mt = 0; mt < 2; ++mt) {
#pragma unroll
    for (int nt = 0; nt < 4; ++nt) {
      int n = col0 + nt * 16 + (lane & 15);
      float bn = bias[n];
#pragma unroll
      for (int v = 0; v < 8; ++v) {
        int m = row0 + mt * 16 + hi * 8 + v;
        out[(size_t)m * CDIM + n] = acc[mt][nt][v] + bn;
      }
    }
  }
}

// ---------------- flash attention: one wave per 16-row Q tile ----------------
// Q,K: [B,H,T,D] f16. Vt: [B,H,D,T] f16. Y: [B*T, C] f16.
__global__ __launch_bounds__(256) void attn_kernel(
    const _Float16* __restrict__ Q, const _Float16* __restrict__ Kh,
    const _Float16* __restrict__ Vt, _Float16* __restrict__ Y) {
  __shared__ __attribute__((aligned(16))) _Float16 pbuf[8][16][40];
  int wid = threadIdx.x >> 5, lane = threadIdx.x & 31;
  int gw = blockIdx.x * 8 + wid;        // global wave id, 8192 total
  int qt = gw & 127;                    // T/16 = 128 Q tiles
  int bh = gw >> 7;                     // b*H + h
  int qb = qt * 16;
  const _Float16* qp = Q + ((size_t)bh * TSEQ + qb) * DH;
  const _Float16* kp = Kh + (size_t)bh * TSEQ * DH;
  const _Float16* vp = Vt + (size_t)bh * DH * TSEQ;

  v16h aq0 = load_a_frag(qp, DH);       // d = 0..31
  v16h aq1 = load_a_frag(qp + 32, DH);  // d = 32..63

  v8f accy0 = {}, accy1 = {}, accy2 = {}, accy3 = {};
  float mrow[8], lrow[8];
#pragma unroll
  for (int v = 0; v < 8; ++v) {
    mrow[v] = -1e30f;
    lrow[v] = 0.f;
  }
  int hi = lane >> 4;
  int hioff = hi * 8;
  _Float16* myp = &pbuf[wid][0][0];
  const float scale = 0.125f;  // 1/sqrt(64)

  for (int j0 = 0; j0 < qb + 16; j0 += 32) {
    // S tile 16(q) x 32(keys): 4 wmma over d = 0..63
    v8f z = {};
    v8f s0 = wmma_f16(aq0, load_b_frag(kp + (size_t)j0 * DH, DH), z);
    s0 = wmma_f16(aq1, load_b_frag(kp + (size_t)j0 * DH + 32, DH), s0);
    v8f s1 = wmma_f16(aq0, load_b_frag(kp + (size_t)(j0 + 16) * DH, DH), z);
    s1 = wmma_f16(aq1, load_b_frag(kp + (size_t)(j0 + 16) * DH + 32, DH), s1);

    int kj = j0 + (lane & 15);
#pragma unroll
    for (int v = 0; v < 8; ++v) {
      int qi = qb + hioff + v;
      float s0v = s0[v] * scale;
      if (kj > qi) s0v = -1e30f;
      float s1v = s1[v] * scale;
      if (kj + 16 > qi) s1v = -1e30f;
      float mb = rmax16(fmaxf(s0v, s1v));
      float mn = fmaxf(mrow[v], mb);
      float corr = __expf(mrow[v] - mn);
      float p0 = __expf(s0v - mn);
      float p1 = __expf(s1v - mn);
      lrow[v] = lrow[v] * corr + rsum16(p0 + p1);
      mrow[v] = mn;
      accy0[v] *= corr;
      accy1[v] *= corr;
      accy2[v] *= corr;
      accy3[v] *= corr;
      // C-layout -> row-major LDS tile (per-wave region; DS ops are
      // wave-ordered, no barrier needed)
      myp[(hioff + v) * 40 + (lane & 15)] = (_Float16)p0;
      myp[(hioff + v) * 40 + 16 + (lane & 15)] = (_Float16)p1;
    }
    // re-load P as an A fragment (row = lane%16, K-runs at hi*8 / hi*8+16)
    {
      int r = lane & 15;
      int kh2 = hi * 8;
      v8h plo = *(const v8h*)&myp[r * 40 + kh2];
      v8h phi = *(const v8h*)&myp[r * 40 + kh2 + 16];
      v16h pa = __builtin_shufflevector(plo, phi, 0, 1, 2, 3, 4, 5, 6, 7, 8, 9,
                                        10, 11, 12, 13, 14, 15);
      accy0 = wmma_f16(pa, load_b_frag(vp + (size_t)0 * TSEQ + j0, TSEQ), accy0);
      accy1 = wmma_f16(pa, load_b_frag(vp + (size_t)16 * TSEQ + j0, TSEQ), accy1);
      accy2 = wmma_f16(pa, load_b_frag(vp + (size_t)32 * TSEQ + j0, TSEQ), accy2);
      accy3 = wmma_f16(pa, load_b_frag(vp + (size_t)48 * TSEQ + j0, TSEQ), accy3);
    }
  }

  // normalize and store y in [B*T, C] f16 for the output projection
  int b = bh >> 4, h = bh & 15;
  v8f ys[4] = {accy0, accy1, accy2, accy3};
#pragma unroll
  for (int dt = 0; dt < 4; ++dt) {
    int d = dt * 16 + (lane & 15);
#pragma unroll
    for (int v = 0; v < 8; ++v) {
      int t = qb + hioff + v;
      float val = ys[dt][v] / lrow[v];
      Y[((size_t)(b * TSEQ + t)) * CDIM + h * DH + d] = (_Float16)val;
    }
  }
}

extern "C" void kernel_launch(void* const* d_in, const int* in_sizes, int n_in,
                              void* d_out, int out_size, void* d_ws,
                              size_t ws_size, hipStream_t stream) {
  (void)in_sizes;
  (void)n_in;
  (void)out_size;
  (void)ws_size;  // requires >= 88 MB
  const float* x = (const float*)d_in[0];
  const float* Wq = (const float*)d_in[1];
  const float* bq = (const float*)d_in[2];
  const float* Wk = (const float*)d_in[3];
  const float* bk = (const float*)d_in[4];
  const float* Wv = (const float*)d_in[5];
  const float* bv = (const float*)d_in[6];
  const float* Wp = (const float*)d_in[7];
  const float* bp = (const float*)d_in[8];

  const size_t MB = 1ull << 20;
  char* ws = (char*)d_ws;
  _Float16* xh = (_Float16*)(ws + 0 * MB);    // 16 MB
  _Float16* wqh = (_Float16*)(ws + 16 * MB);  // 2 MB
  _Float16* wkh = (_Float16*)(ws + 18 * MB);  // 2 MB
  _Float16* wvh = (_Float16*)(ws + 20 * MB);  // 2 MB
  _Float16* wph = (_Float16*)(ws + 22 * MB);  // 2 MB
  _Float16* qh = (_Float16*)(ws + 24 * MB);   // 16 MB [B,H,T,D]
  _Float16* kh = (_Float16*)(ws + 40 * MB);   // 16 MB [B,H,T,D]
  _Float16* vth = (_Float16*)(ws + 56 * MB);  // 16 MB [B,H,D,T]
  _Float16* yh = (_Float16*)(ws + 72 * MB);   // 16 MB [B*T,C]

  const int NX = BATT * TSEQ * CDIM;  // 8388608
  const int NW = CDIM * CDIM;         // 1048576
  cvt_f32_f16_kernel<<<NX / 1024, 256, 0, stream>>>(x, xh, NX);
  cvt_f32_f16_kernel<<<NW / 1024, 256, 0, stream>>>(Wq, wqh, NW);
  cvt_f32_f16_kernel<<<NW / 1024, 256, 0, stream>>>(Wk, wkh, NW);
  cvt_f32_f16_kernel<<<NW / 1024, 256, 0, stream>>>(Wv, wvh, NW);
  cvt_f32_f16_kernel<<<NW / 1024, 256, 0, stream>>>(Wp, wph, NW);

  dim3 gproj(BATT * TSEQ / 128, CDIM / 128, 1);  // 64 x 8
  proj_qkv_kernel<0><<<gproj, 256, 0, stream>>>(xh, wqh, bq, qh);
  proj_qkv_kernel<0><<<gproj, 256, 0, stream>>>(xh, wkh, bk, kh);
  proj_qkv_kernel<1><<<gproj, 256, 0, stream>>>(xh, wvh, bv, vth);

  int nwaves = BATT * NH * (TSEQ / 16);  // 8192
  attn_kernel<<<nwaves / 8, 256, 0, stream>>>(qh, kh, vth, yh);

  proj_out_kernel<<<gproj, 256, 0, stream>>>(yh, wph, bp, (float*)d_out);
}
